// TransformerBlock_56178172232147
// MI455X (gfx1250) — matrix operand order repaired and checked by reading, not compiled, not run on hardware
//
#include <hip/hip_runtime.h>
#include <hip/hip_bf16.h>
#include <math.h>

// ---------------------------------------------------------------- types
typedef __attribute__((ext_vector_type(16))) __bf16 bf16x16;
typedef __attribute__((ext_vector_type(8)))  __bf16 bf16x8;
typedef __attribute__((ext_vector_type(8)))  float  f32x8;

#define S_LEN 4096
#define D_MODEL 1024
#define N_HEADS 16
#define D_HEAD 64
#define D_FF 4096

#define USE_ASYNC_LDS 1

static __device__ __forceinline__ __bf16 f2bf(float f) { return (__bf16)f; }

// Low 32 bits of a generic pointer into LDS == raw LDS byte offset
// (CDNA5 ISA: LDS aperture; LDS address taken from the low 32 bits).
static __device__ __forceinline__ unsigned lds_off_u32(const void* p) {
  return (unsigned)(unsigned long long)p;
}

// Copy a 128-row x 32-col bf16 tile (8KB) global -> LDS.
// 512 x b128 chunks; 2 chunks per thread (256 threads). Async path uses
// GLOBAL_LOAD_ASYNC_TO_LDS_B128 tracked by ASYNCcnt.
static __device__ __forceinline__ void tile_copy_g2l(const __bf16* __restrict__ g0,
                                                     int ld, __bf16* s0) {
  int t = threadIdx.x;
#pragma unroll
  for (int c = t; c < 512; c += 256) {
    int row = c >> 2;
    int q   = (c & 3) << 3;                     // element offset 0,8,16,24
    const __bf16* gp = g0 + (long)row * ld + q;
    __bf16*       sp = s0 + row * 32 + q;
#if USE_ASYNC_LDS
    unsigned loff = lds_off_u32(sp);
    asm volatile("global_load_async_to_lds_b128 %0, %1, off"
                 :: "v"(loff), "v"(gp) : "memory");
#else
    *(bf16x8*)sp = *(const bf16x8*)gp;
#endif
  }
}

// ---------------------------------------------------------------- fragment loaders
// A-fragment (16x32 bf16): lane = M row, 16 contiguous K per lane half (k = 16h + s), loaded as two 8-elem chunks
static __device__ __forceinline__ bf16x16 load_a_frag(const __bf16* __restrict__ A,
                                                      int row0, int k0, int ld) {
  int lane = threadIdx.x & 31;
  int row  = row0 + (lane & 15);
  int kb   = k0 + ((lane >> 4) << 4);           // 0 or 16
  const __bf16* p = A + (long)row * ld + kb;
  union { bf16x16 v; bf16x8 h[2]; } u;
  u.h[0] = *(const bf16x8*)(p);                 // K kb..kb+7
  u.h[1] = *(const bf16x8*)(p + 8);             // K kb+8..kb+15
  return u.v;
}

// B-fragment (32x16 bf16): lane = N col, 16 contiguous K per lane, from W[N][K]
static __device__ __forceinline__ bf16x16 load_b_frag(const __bf16* __restrict__ W,
                                                      int col0, int k0, int ld) {
  int lane = threadIdx.x & 31;
  int col  = col0 + (lane & 15);
  int kb   = k0 + ((lane >> 4) << 4);           // 0 or 16
  return *(const bf16x16*)(W + (long)col * ld + kb);
}

// ---------------------------------------------------------------- f32 -> bf16 convert
__global__ void f32_to_bf16_k(const float* __restrict__ in, __bf16* __restrict__ out, int n) {
  int i = blockIdx.x * blockDim.x + threadIdx.x;
  int stride = gridDim.x * blockDim.x;
  for (; i < n; i += stride) out[i] = f2bf(in[i]);
}

// ---------------------------------------------------------------- RMSNorm -> bf16
__global__ __launch_bounds__(256) void rmsnorm_bf16_k(const float* __restrict__ x,
                                                      const float* __restrict__ g,
                                                      __bf16* __restrict__ out) {
  int row = blockIdx.x;
  int t = threadIdx.x, lane = t & 31, wid = t >> 5;
  const float4* xr = (const float4*)(x + (long)row * D_MODEL);
  float4 v = xr[t];
  float ss = v.x * v.x + v.y * v.y + v.z * v.z + v.w * v.w;
  for (int m = 1; m < 32; m <<= 1) ss += __shfl_xor(ss, m, 32);
  __shared__ float red[8];
  if (lane == 0) red[wid] = ss;
  __syncthreads();
  if (t == 0) {
    float tot = 0.f;
    for (int i = 0; i < 8; ++i) tot += red[i];
    red[0] = tot;
  }
  __syncthreads();
  float rms = rsqrtf(red[0] * (1.0f / (float)D_MODEL) + 1e-5f);
  float4 gv = ((const float4*)g)[t];
  __bf16* o = out + (long)row * D_MODEL + t * 4;
  o[0] = f2bf(v.x * rms * gv.x);
  o[1] = f2bf(v.y * rms * gv.y);
  o[2] = f2bf(v.z * rms * gv.z);
  o[3] = f2bf(v.w * rms * gv.w);
}

// ---------------------------------------------------------------- WMMA GEMM: C = A[M,K] * W[N,K]^T
// Block tile 128x128, 8 waves as 2(M) x 4(N); wave tile 64x32 = 4x2 WMMA tiles.
// K-tiles (128x32 of A and of W) double-buffered in LDS via async global->LDS copies.
// EPI 0: bf16 (scaled); 1: gelu->bf16; 2: +residual->f32; 3: bf16 transposed [N][M]
template <int EPI>
__global__ __launch_bounds__(256) void gemm_bt_k(const __bf16* __restrict__ A,
                                                 const __bf16* __restrict__ W,
                                                 void* __restrict__ out,
                                                 const float* __restrict__ res,
                                                 int M, int N, int K, float scale) {
  __shared__ __align__(16) __bf16 asmem[2][128 * 32];
  __shared__ __align__(16) __bf16 bsmem[2][128 * 32];

  int lane = threadIdx.x & 31;
  int wid  = threadIdx.x >> 5;
  int wm   = wid >> 2;                 // 0..1
  int wn   = wid & 3;                  // 0..3
  int mblk = blockIdx.y * 128;
  int nblk = blockIdx.x * 128;
  int m0l  = wm * 64;
  int n0l  = wn * 32;

  f32x8 acc[4][2];
  f32x8 z = {};
#pragma unroll
  for (int i = 0; i < 4; ++i) { acc[i][0] = z; acc[i][1] = z; }

  int nk = K >> 5;
  tile_copy_g2l(A + (long)mblk * K, K, &asmem[0][0]);
  tile_copy_g2l(W + (long)nblk * K, K, &bsmem[0][0]);

  int coll  = lane & 15;
  int kbA   = (lane >> 4) << 4;        // 0 or 16
  int kbB   = (lane >> 4) << 4;        // 0 or 16

  for (int kt = 0; kt < nk; ++kt) {
    int cur = kt & 1;
    if (kt + 1 < nk) {
      tile_copy_g2l(A + (long)mblk * K + (kt + 1) * 32, K, &asmem[cur ^ 1][0]);
      tile_copy_g2l(W + (long)nblk * K + (kt + 1) * 32, K, &bsmem[cur ^ 1][0]);
#if USE_ASYNC_LDS
      asm volatile("s_wait_asynccnt 4" ::: "memory");
#endif
    } else {
#if USE_ASYNC_LDS
      asm volatile("s_wait_asynccnt 0" ::: "memory");
#endif
    }
    __syncthreads();

    // fragments from LDS
    bf16x16 af[4];
#pragma unroll
    for (int ti = 0; ti < 4; ++ti) {
      const __bf16* p = &asmem[cur][(m0l + ti * 16 + coll) * 32 + kbA];
      union { bf16x16 v; bf16x8 h[2]; } u;
      u.h[0] = *(const bf16x8*)(p);
      u.h[1] = *(const bf16x8*)(p + 8);
      af[ti] = u.v;
    }
    bf16x16 bfr[2];
#pragma unroll
    for (int tj = 0; tj < 2; ++tj)
      bfr[tj] = *(const bf16x16*)(&bsmem[cur][(n0l + tj * 16 + coll) * 32 + kbB]);

#pragma unroll
    for (int ti = 0; ti < 4; ++ti) {
#pragma unroll
      for (int tj = 0; tj < 2; ++tj)
        acc[ti][tj] = __builtin_amdgcn_wmma_f32_16x16x32_bf16(
            false, af[ti], false, bfr[tj], (short)0, acc[ti][tj], false, false);
    }
    __syncthreads();
  }

  int half8 = (lane >> 4) << 3;
#pragma unroll
  for (int ti = 0; ti < 4; ++ti) {
#pragma unroll
    for (int tj = 0; tj < 2; ++tj) {
#pragma unroll
      for (int r = 0; r < 8; ++r) {
        int row = mblk + m0l + ti * 16 + r + half8;
        int col = nblk + n0l + tj * 16 + coll;
        float v = acc[ti][tj][r] * scale;
        if (EPI == 0) {
          ((__bf16*)out)[(long)row * N + col] = f2bf(v);
        } else if (EPI == 1) {
          float gl = 0.5f * v * (1.0f + erff(v * 0.70710678118654752f));
          ((__bf16*)out)[(long)row * N + col] = f2bf(gl);
        } else if (EPI == 2) {
          ((float*)out)[(long)row * N + col] = v + res[(long)row * N + col];
        } else {  // EPI == 3: transposed bf16 store, out[col][row]
          ((__bf16*)out)[(long)col * M + row] = f2bf(v);
        }
      }
    }
  }
}

// ---------------------------------------------------------------- flash attention
// q,k: bf16 [S][D_MODEL] (q pre-scaled by 1/sqrt(d));  vt: bf16 [D_MODEL][S]
// ao: bf16 [S][D_MODEL].  One wave per 16 query rows per head.
__global__ __launch_bounds__(256) void attention_k(const __bf16* __restrict__ q,
                                                   const __bf16* __restrict__ k,
                                                   const __bf16* __restrict__ vt,
                                                   __bf16* __restrict__ ao) {
  __shared__ __align__(32) __bf16 plds[8 * 16 * 32];   // per-wave 16x32 bf16 prob tile
  int lane = threadIdx.x & 31;
  int wid  = threadIdx.x >> 5;
  int head = blockIdx.y;
  int q0   = (blockIdx.x * 8 + wid) * 16;
  int hoff = head * D_HEAD;
  __bf16* myp = &plds[wid * 512];

  bf16x16 qf0 = load_a_frag(q, q0, hoff,      D_MODEL);
  bf16x16 qf1 = load_a_frag(q, q0, hoff + 32, D_MODEL);

  f32x8 z = {};
  f32x8 oacc[4]; oacc[0] = z; oacc[1] = z; oacc[2] = z; oacc[3] = z;
  f32x8 rowmax, rowsum;
  for (int r = 0; r < 8; ++r) { rowmax[r] = -3.0e38f; rowsum[r] = 0.f; }

  int half8 = (lane >> 4) << 3;
  int coll  = lane & 15;
  int nblk  = (q0 + 16 + 31) >> 5;     // causal: kv <= q0+15

  for (int jb = 0; jb < nblk; ++jb) {
    int kv0 = jb << 5;
    f32x8 s0 = z, s1 = z;
    {
      bf16x16 kb0 = load_b_frag(k, kv0,      hoff,      D_MODEL);
      bf16x16 kb1 = load_b_frag(k, kv0,      hoff + 32, D_MODEL);
      bf16x16 kb2 = load_b_frag(k, kv0 + 16, hoff,      D_MODEL);
      bf16x16 kb3 = load_b_frag(k, kv0 + 16, hoff + 32, D_MODEL);
      s0 = __builtin_amdgcn_wmma_f32_16x16x32_bf16(false, qf0, false, kb0, (short)0, s0, false, false);
      s0 = __builtin_amdgcn_wmma_f32_16x16x32_bf16(false, qf1, false, kb1, (short)0, s0, false, false);
      s1 = __builtin_amdgcn_wmma_f32_16x16x32_bf16(false, qf0, false, kb2, (short)0, s1, false, false);
      s1 = __builtin_amdgcn_wmma_f32_16x16x32_bf16(false, qf1, false, kb3, (short)0, s1, false, false);
    }
    if (kv0 + 31 > q0) {                    // causal mask on diagonal blocks
      for (int r = 0; r < 8; ++r) {
        int row = q0 + r + half8;
        if (kv0 + coll      > row) s0[r] = -1.0e30f;
        if (kv0 + 16 + coll > row) s1[r] = -1.0e30f;
      }
    }
    // online softmax (fp32)
    f32x8 bm;
    for (int r = 0; r < 8; ++r) bm[r] = fmaxf(s0[r], s1[r]);
    for (int m = 1; m < 16; m <<= 1)
      for (int r = 0; r < 8; ++r) bm[r] = fmaxf(bm[r], __shfl_xor(bm[r], m, 32));
    f32x8 sc;
    for (int r = 0; r < 8; ++r) {
      float nm = fmaxf(rowmax[r], bm[r]);
      sc[r] = __expf(rowmax[r] - nm);
      rowmax[r] = nm;
      s0[r] = __expf(s0[r] - nm);
      s1[r] = __expf(s1[r] - nm);
    }
    f32x8 ps;
    for (int r = 0; r < 8; ++r) ps[r] = s0[r] + s1[r];
    for (int m = 1; m < 16; m <<= 1)
      for (int r = 0; r < 8; ++r) ps[r] += __shfl_xor(ps[r], m, 32);
    for (int r = 0; r < 8; ++r) rowsum[r] = rowsum[r] * sc[r] + ps[r];
    for (int t = 0; t < 4; ++t) oacc[t] = oacc[t] * sc;

    // transpose P (C layout -> A layout) through LDS as bf16
    for (int r = 0; r < 8; ++r) {
      int row = r + half8;
      myp[row * 32 + coll]      = f2bf(s0[r]);
      myp[row * 32 + 16 + coll] = f2bf(s1[r]);
    }
    asm volatile("s_wait_dscnt 0" ::: "memory");
    bf16x16 pf;
    {
      union { bf16x16 v; bf16x8 h[2]; } u;
      u.h[0] = *(const bf16x8*)(myp + coll * 32 + 2 * half8);
      u.h[1] = *(const bf16x8*)(myp + coll * 32 + 2 * half8 + 8);
      pf = u.v;
    }
    // P x V : vt rows [head*64+d][S] contiguous over kv
    for (int t2 = 0; t2 < 4; ++t2) {
      const __bf16* vp = vt + (long)(hoff + t2 * 16 + coll) * S_LEN + kv0 + ((lane >> 4) << 4);
      bf16x16 vb = *(const bf16x16*)vp;
      oacc[t2] = __builtin_amdgcn_wmma_f32_16x16x32_bf16(false, pf, false, vb, (short)0, oacc[t2], false, false);
    }
  }

  f32x8 inv;
  for (int r = 0; r < 8; ++r) inv[r] = 1.0f / rowsum[r];
  for (int t2 = 0; t2 < 4; ++t2) {
    for (int r = 0; r < 8; ++r) {
      int row = q0 + r + half8;
      ao[(long)row * D_MODEL + hoff + t2 * 16 + coll] = f2bf(oacc[t2][r] * inv[r]);
    }
  }
}

// ---------------------------------------------------------------- launcher
extern "C" void kernel_launch(void* const* d_in, const int* in_sizes, int n_in,
                              void* d_out, int out_size, void* d_ws, size_t ws_size,
                              hipStream_t stream) {
  const float* x  = (const float*)d_in[0];
  const float* wq = (const float*)d_in[1];
  const float* wk = (const float*)d_in[2];
  const float* wv = (const float*)d_in[3];
  const float* wo = (const float*)d_in[4];
  const float* w1 = (const float*)d_in[5];
  const float* w2 = (const float*)d_in[6];
  const float* g1 = (const float*)d_in[7];
  const float* g2 = (const float*)d_in[8];

  const size_t MB = 1u << 20;
  char* ws = (char*)d_ws;
  __bf16* wqb = (__bf16*)(ws + 0 * MB);    // 2 MB each
  __bf16* wkb = (__bf16*)(ws + 2 * MB);
  __bf16* wvb = (__bf16*)(ws + 4 * MB);
  __bf16* wob = (__bf16*)(ws + 6 * MB);
  __bf16* w1b = (__bf16*)(ws + 8 * MB);    // 8 MB
  __bf16* w2b = (__bf16*)(ws + 16 * MB);   // 8 MB
  __bf16* nx  = (__bf16*)(ws + 24 * MB);   // 8 MB
  __bf16* qb  = (__bf16*)(ws + 32 * MB);   // 8 MB
  __bf16* kb  = (__bf16*)(ws + 40 * MB);   // 8 MB
  __bf16* vtb = (__bf16*)(ws + 48 * MB);   // 8 MB (transposed [D][S])
  __bf16* aob = (__bf16*)(ws + 56 * MB);   // 8 MB
  float*  h   = (float*) (ws + 64 * MB);   // 16 MB
  __bf16* nh  = (__bf16*)(ws + 80 * MB);   // 8 MB
  __bf16* tb  = (__bf16*)(ws + 88 * MB);   // 32 MB  -> 120 MB total

  const int S = S_LEN, D = D_MODEL, F = D_FF;

  f32_to_bf16_k<<<512, 256, 0, stream>>>(wq, wqb, D * D);
  f32_to_bf16_k<<<512, 256, 0, stream>>>(wk, wkb, D * D);
  f32_to_bf16_k<<<512, 256, 0, stream>>>(wv, wvb, D * D);
  f32_to_bf16_k<<<512, 256, 0, stream>>>(wo, wob, D * D);
  f32_to_bf16_k<<<2048, 256, 0, stream>>>(w1, w1b, F * D);
  f32_to_bf16_k<<<2048, 256, 0, stream>>>(w2, w2b, F * D);

  rmsnorm_bf16_k<<<S, 256, 0, stream>>>(x, g1, nx);

  gemm_bt_k<0><<<dim3(D / 128, S / 128), 256, 0, stream>>>(nx, wqb, qb,  nullptr, S, D, D, 0.125f);
  gemm_bt_k<0><<<dim3(D / 128, S / 128), 256, 0, stream>>>(nx, wkb, kb,  nullptr, S, D, D, 1.0f);
  gemm_bt_k<3><<<dim3(D / 128, S / 128), 256, 0, stream>>>(nx, wvb, vtb, nullptr, S, D, D, 1.0f);

  attention_k<<<dim3(S / 128, N_HEADS), 256, 0, stream>>>(qb, kb, vtb, aob);

  gemm_bt_k<2><<<dim3(D / 128, S / 128), 256, 0, stream>>>(aob, wob, (void*)h, x, S, D, D, 1.0f);

  rmsnorm_bf16_k<<<S, 256, 0, stream>>>(h, g2, nh);

  gemm_bt_k<1><<<dim3(F / 128, S / 128), 256, 0, stream>>>(nh, w1b, tb, nullptr, S, F, D, 1.0f);

  gemm_bt_k<2><<<dim3(D / 128, S / 128), 256, 0, stream>>>(tb, w2b, d_out, h, S, D, F, 1.0f);
}
